// MoE_15152644620517
// MI455X (gfx1250) — matrix operand order repaired and checked by reading, not compiled, not run on hardware
//
#include <hip/hip_runtime.h>
#include <math.h>

// Problem constants (reference: T,H,I,E,K = 8192,1024,2048,8,2)
#define T_DIM 8192
#define H_DIM 1024
#define I_DIM 2048
#define E_DIM 8
#define K_TOP 2
#define TK (T_DIM * K_TOP)
#define LDK 40  // padded k-extent of LDS tiles (keeps 16B alignment, 40*2=80B rows)

typedef __attribute__((ext_vector_type(16))) __bf16 bf16x16;
typedef __attribute__((ext_vector_type(8)))  __bf16 bf16x8;
typedef __attribute__((ext_vector_type(8)))  float  f32x8;

union Frag16 { bf16x16 v; bf16x8 h[2]; };

__device__ __forceinline__ bf16x8 cvt8v(const float4 a, const float4 b) {
  bf16x8 v;
  v[0] = (__bf16)a.x; v[1] = (__bf16)a.y; v[2] = (__bf16)a.z; v[3] = (__bf16)a.w;
  v[4] = (__bf16)b.x; v[5] = (__bf16)b.y; v[6] = (__bf16)b.z; v[7] = (__bf16)b.w;
  return v;
}

// ---------------------------------------------------------------- zero init
__global__ void moe_zero(float* __restrict__ out, int* __restrict__ counts) {
  size_t i = (size_t)blockIdx.x * blockDim.x + threadIdx.x;
  const size_t n = (size_t)T_DIM * H_DIM;
  const size_t stride = (size_t)gridDim.x * blockDim.x;
  for (; i < n; i += stride) out[i] = 0.0f;
  if (blockIdx.x == 0 && threadIdx.x < E_DIM) counts[threadIdx.x] = 0;
}

// ---------------------------------------------------------------- router
// One wave32 per token: logits = x[t] @ Wr + br, top-2, sigmoid gate renorm.
__global__ __launch_bounds__(256) void moe_router(
    const float* __restrict__ x, const float* __restrict__ Wr,
    const float* __restrict__ br, float* __restrict__ topg,
    int* __restrict__ topi, int* __restrict__ counts) {
  int wave = threadIdx.x >> 5, lane = threadIdx.x & 31;
  int t = blockIdx.x * 8 + wave;
  float acc[E_DIM];
#pragma unroll
  for (int e = 0; e < E_DIM; ++e) acc[e] = 0.0f;
  const float* xr = x + (size_t)t * H_DIM;
  for (int j = 0; j < H_DIM / 32; ++j) {
    int k = j * 32 + lane;
    float xv = xr[k];
    const float* wr = Wr + (size_t)k * E_DIM;  // contiguous 8 floats per lane
#pragma unroll
    for (int e = 0; e < E_DIM; ++e) acc[e] += xv * wr[e];
  }
#pragma unroll
  for (int e = 0; e < E_DIM; ++e) {
#pragma unroll
    for (int m = 16; m >= 1; m >>= 1) acc[e] += __shfl_xor(acc[e], m, 32);
  }
  if (lane == 0) {
    float v0 = -3.4e38f, v1 = -3.4e38f;
    int i0 = 0, i1 = 0;
#pragma unroll
    for (int e = 0; e < E_DIM; ++e) {
      float l = acc[e] + br[e];
      if (l > v0) { v1 = v0; i1 = i0; v0 = l; i0 = e; }
      else if (l > v1) { v1 = l; i1 = e; }
    }
    float g0 = __builtin_amdgcn_rcpf(1.0f + __expf(-v0));
    float g1 = __builtin_amdgcn_rcpf(1.0f + __expf(-v1));
    float rs = __builtin_amdgcn_rcpf(g0 + g1 + 1e-10f);
    topg[t * 2 + 0] = g0 * rs;
    topg[t * 2 + 1] = g1 * rs;
    topi[t * 2 + 0] = i0;
    topi[t * 2 + 1] = i1;
    atomicAdd(&counts[i0], 1);
    atomicAdd(&counts[i1], 1);
  }
}

// ---------------------------------------------------------------- scan (E=8)
__global__ void moe_scan(const int* __restrict__ counts, int* __restrict__ offsets,
                         int* __restrict__ cursors) {
  if (threadIdx.x == 0) {
    int o = 0;
    for (int e = 0; e < E_DIM; ++e) { offsets[e] = o; o += counts[e]; cursors[e] = 0; }
    offsets[E_DIM] = o;  // == TK
  }
}

// ---------------------------------------------------------------- assign
__global__ __launch_bounds__(256) void moe_assign(
    const int* __restrict__ topi, const float* __restrict__ topg,
    const int* __restrict__ offsets, int* __restrict__ cursors,
    int* __restrict__ slot_token, float* __restrict__ slot_gate) {
  int t = blockIdx.x * blockDim.x + threadIdx.x;
  if (t >= T_DIM) return;
#pragma unroll
  for (int kk = 0; kk < K_TOP; ++kk) {
    int e = topi[t * 2 + kk];
    int pos = atomicAdd(&cursors[e], 1);
    int s = offsets[e] + pos;
    slot_token[s] = t;
    slot_gate[s] = topg[t * 2 + kk];
  }
}

// ---------------------------------------------------------------- GEMM1
// h[slot, I] = silu(x[tok] @ Wg[e]) * (x[tok] @ Wu[e]) * gate, bf16 WMMA.
// Block: 128 thr (4 waves, 2x2). Block tile M=64 x N=64; wave tile 32x32.
// Register-staged pipeline: next k-slice prefetched (and cvt'd) into VGPRs
// during the WMMA phase; ds_store at the top of the next iteration.
__global__ __launch_bounds__(128) void moe_gemm1(
    const float* __restrict__ x, const float* __restrict__ Wg,
    const float* __restrict__ Wu, const int* __restrict__ slot_token,
    const float* __restrict__ slot_gate, const int* __restrict__ offsets,
    __bf16* __restrict__ hbuf) {
  __shared__ __align__(16) __bf16 Alds[64 * LDK];   // row-major [m][k]
  __shared__ __align__(16) __bf16 Bgls[64 * LDK];   // transposed [n][k]
  __shared__ __align__(16) __bf16 Buls[64 * LDK];   // transposed [n][k]
  __shared__ float gate_s[64];
  __shared__ int tok_s[64];

  const int e = blockIdx.z;
  const int base = offsets[e];
  const int cnt = offsets[e + 1] - base;
  const int m0 = blockIdx.y * 64;
  if (m0 >= cnt) return;
  const int n0 = blockIdx.x * 64;
  const int tid = threadIdx.x;
  const int wave = tid >> 5, lane = tid & 31;
  const int nlane = lane & 15, asel = lane >> 4;
  const int wave_m = wave >> 1, wave_n = wave & 1;

  if (tid < 64) {
    int m = m0 + tid;
    bool valid = m < cnt;
    tok_s[tid] = valid ? slot_token[base + m] : 0;
    gate_s[tid] = valid ? slot_gate[base + m] : 0.0f;
  }
  __syncthreads();

  const float* WgE = Wg + (size_t)e * H_DIM * I_DIM;
  const float* WuE = Wu + (size_t)e * H_DIM * I_DIM;

  // Per-thread k-invariant staging descriptors.
  const float* asrc[2];
  __bf16* adst[2];
#pragma unroll
  for (int i = 0; i < 2; ++i) {
    int run = tid + i * 128;           // 256 runs: 64 rows x 4 ksegs
    int row = run >> 2, kseg = run & 3;
    asrc[i] = x + (size_t)tok_s[row] * H_DIM + kseg * 8;
    adst[i] = &Alds[row * LDK + kseg * 8];
  }
  const float* bsrc[4];
  __bf16* bdst[4];
#pragma unroll
  for (int i = 0; i < 4; ++i) {
    int run = tid + i * 128;           // 512 runs: 2 mats x 64 cols x 4 ksegs
    int mat = run >> 8, rem = run & 255;
    int kseg = rem >> 6, col = rem & 63;
    bsrc[i] = (mat ? WuE : WgE) + (size_t)(kseg * 8) * I_DIM + n0 + col;
    bdst[i] = (mat ? Buls : Bgls) + col * LDK + kseg * 8;
  }
  // k-invariant LDS fragment pointers.
  const __bf16* apF[2];
  const __bf16* bgF[2];
  const __bf16* buF[2];
#pragma unroll
  for (int i = 0; i < 2; ++i) {
    apF[i] = &Alds[(wave_m * 32 + i * 16 + nlane) * LDK + asel * 8];
    bgF[i] = &Bgls[(wave_n * 32 + i * 16 + nlane) * LDK + asel * 8];
    buF[i] = &Buls[(wave_n * 32 + i * 16 + nlane) * LDK + asel * 8];
  }

  bf16x8 areg[2], breg[4];
  auto load_slice = [&](int k0) {
#pragma unroll
    for (int i = 0; i < 2; ++i) {
      const float* s = asrc[i] + k0;
      float4 f0 = *(const float4*)s;
      float4 f1 = *(const float4*)(s + 4);
      areg[i] = cvt8v(f0, f1);
    }
#pragma unroll
    for (int i = 0; i < 4; ++i) {
      const float* s = bsrc[i] + (size_t)k0 * I_DIM;
      float tv[8];
#pragma unroll
      for (int j = 0; j < 8; ++j) tv[j] = s[(size_t)j * I_DIM];
      bf16x8 v;
#pragma unroll
      for (int j = 0; j < 8; ++j) v[j] = (__bf16)tv[j];
      breg[i] = v;
    }
  };

  f32x8 accg[2][2], accu[2][2];
#pragma unroll
  for (int mi = 0; mi < 2; ++mi)
#pragma unroll
    for (int ni = 0; ni < 2; ++ni)
#pragma unroll
      for (int r = 0; r < 8; ++r) { accg[mi][ni][r] = 0.0f; accu[mi][ni][r] = 0.0f; }

  load_slice(0);
  const int NK = H_DIM / 32;
  for (int kt = 0; kt < NK; ++kt) {
    // Commit the prefetched slice to LDS.
#pragma unroll
    for (int i = 0; i < 2; ++i) *(bf16x8*)adst[i] = areg[i];
#pragma unroll
    for (int i = 0; i < 4; ++i) *(bf16x8*)bdst[i] = breg[i];
    __syncthreads();
    // Kick off next slice's global loads; latency hidden by WMMA phase.
    if (kt + 1 < NK) load_slice((kt + 1) * 32);

    Frag16 af[2];
#pragma unroll
    for (int mi = 0; mi < 2; ++mi) {
      af[mi].h[0] = *(const bf16x8*)apF[mi];
      af[mi].h[1] = *(const bf16x8*)(apF[mi] + 16);
    }
#pragma unroll
    for (int ni = 0; ni < 2; ++ni) {
      Frag16 bg;
      bg.h[0] = *(const bf16x8*)bgF[ni];
      bg.h[1] = *(const bf16x8*)(bgF[ni] + 16);
#pragma unroll
      for (int mi = 0; mi < 2; ++mi)
        accg[mi][ni] = __builtin_amdgcn_wmma_f32_16x16x32_bf16(
            false, af[mi].v, false, bg.v, (short)0, accg[mi][ni], false, false);
      Frag16 bu;
      bu.h[0] = *(const bf16x8*)buF[ni];
      bu.h[1] = *(const bf16x8*)(buF[ni] + 16);
#pragma unroll
      for (int mi = 0; mi < 2; ++mi)
        accu[mi][ni] = __builtin_amdgcn_wmma_f32_16x16x32_bf16(
            false, af[mi].v, false, bu.v, (short)0, accu[mi][ni], false, false);
    }
    __syncthreads();
  }

  // Epilogue: h = silu(g) * u * gate -> bf16 hbuf (v_rcp_f32 sigmoid).
#pragma unroll
  for (int mi = 0; mi < 2; ++mi)
#pragma unroll
    for (int ni = 0; ni < 2; ++ni) {
      int n = n0 + wave_n * 32 + ni * 16 + nlane;
#pragma unroll
      for (int r = 0; r < 8; ++r) {
        int lrow = wave_m * 32 + mi * 16 + asel * 8 + r;
        int m = m0 + lrow;
        if (m < cnt) {
          float g = accg[mi][ni][r], u = accu[mi][ni][r];
          float sig = __builtin_amdgcn_rcpf(1.0f + __expf(-g));
          float hv = g * sig * u * gate_s[lrow];
          hbuf[(size_t)(base + m) * I_DIM + n] = (__bf16)hv;
        }
      }
    }
}

// ---------------------------------------------------------------- GEMM2
// out[tok] += h[slot] @ Wd[e].  A already bf16 (direct b128 loads).
__global__ __launch_bounds__(128) void moe_gemm2(
    const __bf16* __restrict__ hbuf, const float* __restrict__ Wd,
    const int* __restrict__ slot_token, const int* __restrict__ offsets,
    float* __restrict__ out) {
  __shared__ __align__(16) __bf16 Blds[64 * LDK];  // transposed [n][k]
  __shared__ int tok_s[64];

  const int e = blockIdx.z;
  const int base = offsets[e];
  const int cnt = offsets[e + 1] - base;
  const int m0 = blockIdx.y * 64;
  if (m0 >= cnt) return;
  const int n0 = blockIdx.x * 64;
  const int tid = threadIdx.x;
  const int wave = tid >> 5, lane = tid & 31;
  const int nlane = lane & 15, asel = lane >> 4;
  const int wave_m = wave >> 1, wave_n = wave & 1;

  if (tid < 64) {
    int m = m0 + tid;
    tok_s[tid] = (m < cnt) ? slot_token[base + m] : 0;
  }
  __syncthreads();

  const float* WdE = Wd + (size_t)e * I_DIM * H_DIM;

  // A rows direct from global hbuf (already bf16, 16B-aligned runs).
  const __bf16* arp[2];
#pragma unroll
  for (int mi = 0; mi < 2; ++mi) {
    int r = m0 + wave_m * 32 + mi * 16 + nlane;
    if (r >= cnt) r = cnt - 1;  // clamp: garbage rows never stored
    arp[mi] = hbuf + (size_t)(base + r) * I_DIM + asel * 8;
  }
  // B staging descriptors (256 runs: 64 cols x 4 ksegs).
  const float* bsrc[2];
  __bf16* bdst[2];
#pragma unroll
  for (int i = 0; i < 2; ++i) {
    int run = tid + i * 128;
    int kseg = run >> 6, col = run & 63;
    bsrc[i] = WdE + (size_t)(kseg * 8) * H_DIM + n0 + col;
    bdst[i] = &Blds[col * LDK + kseg * 8];
  }
  const __bf16* bpF[2];
#pragma unroll
  for (int i = 0; i < 2; ++i)
    bpF[i] = &Blds[(wave_n * 32 + i * 16 + nlane) * LDK + asel * 8];

  bf16x8 breg[2];
  auto load_slice = [&](int k0) {
#pragma unroll
    for (int i = 0; i < 2; ++i) {
      const float* s = bsrc[i] + (size_t)k0 * H_DIM;
      float tv[8];
#pragma unroll
      for (int j = 0; j < 8; ++j) tv[j] = s[(size_t)j * H_DIM];
      bf16x8 v;
#pragma unroll
      for (int j = 0; j < 8; ++j) v[j] = (__bf16)tv[j];
      breg[i] = v;
    }
  };

  f32x8 acc[2][2];
#pragma unroll
  for (int mi = 0; mi < 2; ++mi)
#pragma unroll
    for (int ni = 0; ni < 2; ++ni)
#pragma unroll
      for (int r = 0; r < 8; ++r) acc[mi][ni][r] = 0.0f;

  load_slice(0);
  const int NK = I_DIM / 32;
  for (int kt = 0; kt < NK; ++kt) {
    int k0 = kt * 32;
#pragma unroll
    for (int i = 0; i < 2; ++i) *(bf16x8*)bdst[i] = breg[i];
    __syncthreads();
    if (kt + 1 < NK) load_slice(k0 + 32);

    Frag16 af[2];
#pragma unroll
    for (int mi = 0; mi < 2; ++mi) {
      af[mi].h[0] = *(const bf16x8*)(arp[mi] + k0);
      af[mi].h[1] = *(const bf16x8*)(arp[mi] + k0 + 16);
    }
#pragma unroll
    for (int ni = 0; ni < 2; ++ni) {
      Frag16 bfr;
      bfr.h[0] = *(const bf16x8*)bpF[ni];
      bfr.h[1] = *(const bf16x8*)(bpF[ni] + 16);
#pragma unroll
      for (int mi = 0; mi < 2; ++mi)
        acc[mi][ni] = __builtin_amdgcn_wmma_f32_16x16x32_bf16(
            false, af[mi].v, false, bfr.v, (short)0, acc[mi][ni], false, false);
    }
    __syncthreads();
  }

  // Scatter-accumulate: exactly 2 contributions per out element (commutative
  // IEEE adds onto 0 -> bitwise deterministic).
#pragma unroll
  for (int mi = 0; mi < 2; ++mi)
#pragma unroll
    for (int ni = 0; ni < 2; ++ni) {
      int n = n0 + wave_n * 32 + ni * 16 + nlane;
#pragma unroll
      for (int r = 0; r < 8; ++r) {
        int lrow = wave_m * 32 + mi * 16 + asel * 8 + r;
        int m = m0 + lrow;
        if (m < cnt) {
          float* p = out + (size_t)tok_s[lrow] * H_DIM + n;
          __hip_atomic_fetch_add(p, acc[mi][ni][r], __ATOMIC_RELAXED,
                                 __HIP_MEMORY_SCOPE_AGENT);
        }
      }
    }
}

// ---------------------------------------------------------------- launch
extern "C" void kernel_launch(void* const* d_in, const int* in_sizes, int n_in,
                              void* d_out, int out_size, void* d_ws,
                              size_t ws_size, hipStream_t stream) {
  (void)in_sizes; (void)n_in; (void)out_size; (void)ws_size;
  const float* x  = (const float*)d_in[0];
  const float* Wr = (const float*)d_in[1];
  const float* br = (const float*)d_in[2];
  const float* Wg = (const float*)d_in[3];
  const float* Wu = (const float*)d_in[4];
  const float* Wd = (const float*)d_in[5];
  float* out = (float*)d_out;

  // Workspace layout (~64.25 MB total)
  char* w = (char*)d_ws;
  __bf16* hbuf = (__bf16*)w;                 size_t off = (size_t)TK * I_DIM * 2;   // 64 MB
  float* topg       = (float*)(w + off);     off += (size_t)T_DIM * K_TOP * 4;
  int*   topi       = (int*)(w + off);       off += (size_t)T_DIM * K_TOP * 4;
  int*   slot_token = (int*)(w + off);       off += (size_t)TK * 4;
  float* slot_gate  = (float*)(w + off);     off += (size_t)TK * 4;
  int*   counts     = (int*)(w + off);       off += 256;
  int*   offsets    = (int*)(w + off);       off += 256;
  int*   cursors    = (int*)(w + off);       off += 256;

  moe_zero<<<2048, 256, 0, stream>>>(out, counts);
  moe_router<<<T_DIM / 8, 256, 0, stream>>>(x, Wr, br, topg, topi, counts);
  moe_scan<<<1, 32, 0, stream>>>(counts, offsets, cursors);
  moe_assign<<<T_DIM / 256, 256, 0, stream>>>(topi, topg, offsets, cursors,
                                              slot_token, slot_gate);
  moe_gemm1<<<dim3(I_DIM / 64, T_DIM / 64, E_DIM), 128, 0, stream>>>(
      x, Wg, Wu, slot_token, slot_gate, offsets, hbuf);
  moe_gemm2<<<dim3(H_DIM / 64, T_DIM / 64, E_DIM), 128, 0, stream>>>(
      hbuf, Wd, slot_token, offsets, out);
}
